// Encoder1DBlock_3066606649374
// MI455X (gfx1250) — compile-verified, matrix operand order checked by reading, not run on hardware
//
#include <hip/hip_runtime.h>
#include <hip/hip_bf16.h>
#include <math.h>

// ---------------------------------------------------------------------------
// Problem constants (from reference): N=8, L=1024, H=16, D=64, HD=1024, MLP=4096
// ---------------------------------------------------------------------------
#define NB    8
#define LSEQ  1024
#define NH    16
#define DH    64
#define HDIM  1024
#define MLPD  4096
#define MTOK  (NB * LSEQ)          // 8192 token rows

typedef __bf16 bf16;
typedef __attribute__((ext_vector_type(16))) __bf16 bf16x16;
typedef __attribute__((ext_vector_type(8)))  float  f32x8;
typedef int v4i __attribute__((vector_size(16)));

// ---- CDNA5 async global->LDS copies (ASYNCcnt), guarded by probe -----------
#if defined(__has_builtin)
#  if __has_builtin(__builtin_amdgcn_global_load_async_to_lds_b128) && \
      __has_builtin(__builtin_amdgcn_s_wait_asynccnt)
#    define HAVE_ASYNC_LDS 1
#  endif
#endif
#ifndef HAVE_ASYNC_LDS
#  define HAVE_ASYNC_LDS 0
#endif

#if HAVE_ASYNC_LDS
__device__ __forceinline__ void async_copy_b128(const bf16* g, bf16* l) {
    __builtin_amdgcn_global_load_async_to_lds_b128(
        (__attribute__((address_space(1))) v4i*)g,
        (__attribute__((address_space(3))) v4i*)l, 0, 0);
}
__device__ __forceinline__ void async_wait_all() {
    __builtin_amdgcn_s_wait_asynccnt(0);
}
#endif

__device__ __forceinline__ float gelu_tanh(float x) {
    const float c = 0.7978845608028654f;   // sqrt(2/pi)
    float x3 = x * x * x;
    return 0.5f * x * (1.0f + tanhf(c * (x + 0.044715f * x3)));
}

enum { EPI_BIAS = 0, EPI_BIAS_GELU = 1, EPI_BIAS_RES = 2, EPI_BIAS_VT = 3 };

// ---------------------------------------------------------------------------
// Weight convert + transpose: w (K x N, f32 row-major) -> wt (N x K, bf16)
// ---------------------------------------------------------------------------
__global__ __launch_bounds__(256)
void transpose_to_bf16(const float* __restrict__ w, bf16* __restrict__ wt,
                       int K, int N) {
    __shared__ float t[32][33];
    const int bn = blockIdx.x * 32;
    const int bk = blockIdx.y * 32;
    const int tx = threadIdx.x & 31;
    const int ty = threadIdx.x >> 5;         // 0..7
#pragma unroll
    for (int r = 0; r < 32; r += 8)
        t[ty + r][tx] = w[(size_t)(bk + ty + r) * N + bn + tx];
    __syncthreads();
#pragma unroll
    for (int r = 0; r < 32; r += 8)
        wt[(size_t)(bn + ty + r) * K + bk + tx] = (bf16)t[tx][ty + r];
}

// ---------------------------------------------------------------------------
// LayerNorm (row of 1024 f32) fused with bf16 downconvert. One block per row.
// ---------------------------------------------------------------------------
__global__ __launch_bounds__(256)
void layernorm_to_bf16(const float* __restrict__ x, const float* __restrict__ g,
                       const float* __restrict__ b, bf16* __restrict__ y) {
    __shared__ float red[2][8];
    const int row = blockIdx.x;
    const float* xr = x + (size_t)row * HDIM;
    float4 v = ((const float4*)xr)[threadIdx.x];
    float s  = v.x + v.y + v.z + v.w;
    float s2 = v.x * v.x + v.y * v.y + v.z * v.z + v.w * v.w;
#pragma unroll
    for (int m = 16; m > 0; m >>= 1) { s += __shfl_xor(s, m); s2 += __shfl_xor(s2, m); }
    const int wave = threadIdx.x >> 5, lane = threadIdx.x & 31;
    if (lane == 0) { red[0][wave] = s; red[1][wave] = s2; }
    __syncthreads();
    if (wave == 0) {
        float a  = (lane < 8) ? red[0][lane] : 0.0f;
        float a2 = (lane < 8) ? red[1][lane] : 0.0f;
#pragma unroll
        for (int m = 4; m > 0; m >>= 1) { a += __shfl_xor(a, m); a2 += __shfl_xor(a2, m); }
        if (lane == 0) { red[0][0] = a; red[1][0] = a2; }
    }
    __syncthreads();
    const float mu   = red[0][0] * (1.0f / HDIM);
    const float var  = red[1][0] * (1.0f / HDIM) - mu * mu;
    const float rstd = rsqrtf(var + 1e-5f);
    float4 gv = ((const float4*)g)[threadIdx.x];
    float4 bv = ((const float4*)b)[threadIdx.x];
    bf16* yr = y + (size_t)row * HDIM;
    const int c = threadIdx.x * 4;
    yr[c + 0] = (bf16)((v.x - mu) * rstd * gv.x + bv.x);
    yr[c + 1] = (bf16)((v.y - mu) * rstd * gv.y + bv.y);
    yr[c + 2] = (bf16)((v.z - mu) * rstd * gv.z + bv.z);
    yr[c + 3] = (bf16)((v.w - mu) * rstd * gv.w + bv.w);
}

// ---------------------------------------------------------------------------
// Generic bf16 WMMA GEMM:  C[M,N] = epilogue(A[M,K] * BT[N,K]^T + bias)
// Block tile 128x128, 8 waves, each wave 32x64 (2x4 wmma tiles), K-step 32.
// Double-buffered LDS; async global->LDS prefetch of the next K tile when the
// toolchain exposes GLOBAL_LOAD_ASYNC_TO_LDS_B128.
// EPI_BIAS_VT: scatter-store V into VT[batch][head][d][key] for attention.
// ---------------------------------------------------------------------------
template <int EPI, typename OutT>
__global__ __launch_bounds__(256)
void gemm_bf16_wmma(const bf16* __restrict__ A, const bf16* __restrict__ BT,
                    const float* __restrict__ bias, const float* __restrict__ res,
                    OutT* __restrict__ C, int M, int N, int K) {
    __shared__ bf16 As[2][128][32];
    __shared__ bf16 Bs[2][128][32];

    const int tid  = threadIdx.x;
    const int wave = tid >> 5;
    const int lane = tid & 31;
    const int lh   = lane >> 4;     // K-half selector (lanes 16-31 hold K=16..31)
    const int ll   = lane & 15;     // row/col within 16
    const int wm   = wave & 3;      // 4 waves along M (32 rows each)
    const int wn   = wave >> 2;     // 2 waves along N (64 cols each)

    const int bm = blockIdx.y * 128;
    const int bn = blockIdx.x * 128;

    f32x8 acc[2][4];
#pragma unroll
    for (int i = 0; i < 2; ++i)
#pragma unroll
        for (int j = 0; j < 4; ++j)
#pragma unroll
            for (int e = 0; e < 8; ++e) acc[i][j][e] = 0.0f;

    const int lrow = tid >> 1;            // 0..127
    const int lcol = (tid & 1) * 16;      // 0 or 16
    const bf16* aRow = A  + (size_t)(bm + lrow) * K + lcol;
    const bf16* bRow = BT + (size_t)(bn + lrow) * K + lcol;

    const int nk = K / 32;

#if HAVE_ASYNC_LDS
    auto issue = [&](int buf, int k0) {
        async_copy_b128(aRow + k0,     &As[buf][lrow][lcol]);
        async_copy_b128(aRow + k0 + 8, &As[buf][lrow][lcol + 8]);
        async_copy_b128(bRow + k0,     &Bs[buf][lrow][lcol]);
        async_copy_b128(bRow + k0 + 8, &Bs[buf][lrow][lcol + 8]);
    };
    issue(0, 0);
    async_wait_all();
    __syncthreads();
#else
    {
        uint4 a0 = *(const uint4*)(aRow);
        uint4 a1 = *(const uint4*)(aRow + 8);
        uint4 b0 = *(const uint4*)(bRow);
        uint4 b1 = *(const uint4*)(bRow + 8);
        *(uint4*)&As[0][lrow][lcol]     = a0;
        *(uint4*)&As[0][lrow][lcol + 8] = a1;
        *(uint4*)&Bs[0][lrow][lcol]     = b0;
        *(uint4*)&Bs[0][lrow][lcol + 8] = b1;
        __syncthreads();
    }
#endif

    for (int kt = 0; kt < nk; ++kt) {
        const int cur = kt & 1;
#if HAVE_ASYNC_LDS
        if (kt + 1 < nk) issue(cur ^ 1, (kt + 1) * 32);   // overlap with compute
#else
        uint4 a0 = {}, a1 = {}, b0 = {}, b1 = {};
        const bool pre = (kt + 1 < nk);
        if (pre) {
            const bf16* an = aRow + (kt + 1) * 32;
            const bf16* bn2 = bRow + (kt + 1) * 32;
            a0 = *(const uint4*)(an);
            a1 = *(const uint4*)(an + 8);
            b0 = *(const uint4*)(bn2);
            b1 = *(const uint4*)(bn2 + 8);
            __builtin_prefetch(an + 32, 0, 0);
            __builtin_prefetch(bn2 + 32, 0, 0);
        }
#endif
        bf16x16 af[2], bfr[4];
#pragma unroll
        for (int i = 0; i < 2; ++i)
            af[i] = *(const bf16x16*)&As[cur][wm * 32 + i * 16 + ll][lh * 16];
#pragma unroll
        for (int j = 0; j < 4; ++j)
            bfr[j] = *(const bf16x16*)&Bs[cur][wn * 64 + j * 16 + ll][lh * 16];

#pragma unroll
        for (int i = 0; i < 2; ++i)
#pragma unroll
            for (int j = 0; j < 4; ++j)
                acc[i][j] = __builtin_amdgcn_wmma_f32_16x16x32_bf16(
                    false, af[i], false, bfr[j], (short)0, acc[i][j], false, false);

#if HAVE_ASYNC_LDS
        async_wait_all();
        __syncthreads();
#else
        __syncthreads();
        if (pre) {
            *(uint4*)&As[cur ^ 1][lrow][lcol]     = a0;
            *(uint4*)&As[cur ^ 1][lrow][lcol + 8] = a1;
            *(uint4*)&Bs[cur ^ 1][lrow][lcol]     = b0;
            *(uint4*)&Bs[cur ^ 1][lrow][lcol + 8] = b1;
        }
        __syncthreads();
#endif
    }

    // Epilogue. C layout: VGPR e -> row = e + lh*8 ; lane%16 -> col.
#pragma unroll
    for (int i = 0; i < 2; ++i) {
#pragma unroll
        for (int j = 0; j < 4; ++j) {
            const int col = bn + wn * 64 + j * 16 + ll;
            const float bvv = bias ? bias[col] : 0.0f;
#pragma unroll
            for (int e = 0; e < 8; ++e) {
                const int row = bm + wm * 32 + i * 16 + lh * 8 + e;
                float v = acc[i][j][e] + bvv;
                if (EPI == EPI_BIAS_GELU) v = gelu_tanh(v);
                if (EPI == EPI_BIAS_RES)  v += res[(size_t)row * N + col];
                if (EPI == EPI_BIAS_VT) {
                    // row = batch*LSEQ + key ; col = head*DH + d
                    const size_t idx =
                        (((size_t)(row >> 10) * NH + (col >> 6)) * DH + (col & (DH - 1)))
                            * LSEQ + (row & (LSEQ - 1));
                    C[idx] = (OutT)v;
                } else {
                    C[(size_t)row * N + col] = (OutT)v;
                }
            }
        }
    }
}

// ---------------------------------------------------------------------------
// Flash-attention per (batch, head, 64-query tile). 128 threads = 4 waves,
// each wave owns 16 query rows. V arrives pre-transposed (VT[b][h][d][key])
// so the P*V B-operand loads straight with 128-bit copies -- no LDS transpose.
// ---------------------------------------------------------------------------
__global__ __launch_bounds__(128)
void attention_wmma(const bf16* __restrict__ Q, const bf16* __restrict__ Km,
                    const bf16* __restrict__ VT, bf16* __restrict__ O) {
    __shared__ bf16 Qs[64][64];          //  8 KB
    __shared__ bf16 Ks[128][64];         // 16 KB
    __shared__ bf16 VTs[64][144];        // 18 KB (pad: rows 32B-aligned)
    __shared__ bf16 Ps[4][16][144];      // 18 KB per-wave P staging

    const int tid  = threadIdx.x;
    const int wave = tid >> 5;
    const int lane = tid & 31;
    const int lh   = lane >> 4;
    const int ll   = lane & 15;

    const int qt    = blockIdx.x;        // 16 query tiles of 64
    const int head  = blockIdx.y;        // 16
    const int batch = blockIdx.z;        // 8
    const size_t base   = ((size_t)batch * LSEQ) * HDIM + (size_t)head * DH;
    const size_t vtbase = ((size_t)(batch * NH + head)) * DH * LSEQ;
    const int q0 = qt * 64;

    // Load Q tile (64 x 64): 128 threads x 32 bf16 each.
    {
        const int row = tid >> 1, half = tid & 1;
        const uint4* s4 = (const uint4*)(Q + base + (size_t)(q0 + row) * HDIM + half * 32);
        uint4* dst = (uint4*)&Qs[row][half * 32];
        dst[0] = s4[0]; dst[1] = s4[1]; dst[2] = s4[2]; dst[3] = s4[3];
    }

    f32x8 acco[4];
#pragma unroll
    for (int j = 0; j < 4; ++j)
#pragma unroll
        for (int e = 0; e < 8; ++e) acco[j][e] = 0.0f;
    float mrow[8], lrow[8];
#pragma unroll
    for (int e = 0; e < 8; ++e) { mrow[e] = -1e30f; lrow[e] = 0.0f; }

    for (int kt = 0; kt < LSEQ / 128; ++kt) {
        const int k0 = kt * 128;
        __syncthreads();   // previous tile fully consumed (also orders Q store)
        {
            // K tile: one full 64-wide row (128 B) per thread.
            const bf16* ks = Km + base + (size_t)(k0 + tid) * HDIM;
            bf16* kd = &Ks[tid][0];
            // VT tile: d = tid/2, 64 keys (128 B) per thread.
            const int d = tid >> 1, half = tid & 1;
            const bf16* vs = VT + vtbase + (size_t)d * LSEQ + k0 + half * 64;
            bf16* vd = &VTs[d][half * 64];
#if HAVE_ASYNC_LDS
#pragma unroll
            for (int c = 0; c < 64; c += 8) async_copy_b128(ks + c, kd + c);
#pragma unroll
            for (int c = 0; c < 64; c += 8) async_copy_b128(vs + c, vd + c);
#else
#pragma unroll
            for (int c = 0; c < 8; ++c) ((uint4*)kd)[c] = ((const uint4*)ks)[c];
#pragma unroll
            for (int c = 0; c < 8; ++c) ((uint4*)vd)[c] = ((const uint4*)vs)[c];
#endif
        }
#if HAVE_ASYNC_LDS
        async_wait_all();
#endif
        __syncthreads();

        // ---- S = (Q K^T) * 1/sqrt(D) ----
        f32x8 accs[8];
#pragma unroll
        for (int jt = 0; jt < 8; ++jt) {
#pragma unroll
            for (int e = 0; e < 8; ++e) accs[jt][e] = 0.0f;
#pragma unroll
            for (int kc = 0; kc < 2; ++kc) {
                bf16x16 a = *(const bf16x16*)&Qs[wave * 16 + ll][kc * 32 + lh * 16];
                bf16x16 b = *(const bf16x16*)&Ks[jt * 16 + ll][kc * 32 + lh * 16];
                accs[jt] = __builtin_amdgcn_wmma_f32_16x16x32_bf16(
                    false, a, false, b, (short)0, accs[jt], false, false);
            }
        }
        const float scale = 0.125f;     // 1/sqrt(64)

        // ---- online softmax: row = (VGPR e, lane-half); cols across 16 lanes
        float mnew[8];
#pragma unroll
        for (int e = 0; e < 8; ++e) {
            float mx = -1e30f;
#pragma unroll
            for (int jt = 0; jt < 8; ++jt) {
                accs[jt][e] *= scale;
                mx = fmaxf(mx, accs[jt][e]);
            }
#pragma unroll
            for (int m = 1; m < 16; m <<= 1) mx = fmaxf(mx, __shfl_xor(mx, m));
            mnew[e] = fmaxf(mrow[e], mx);
        }
#pragma unroll
        for (int e = 0; e < 8; ++e) {
            float sum = 0.0f;
#pragma unroll
            for (int jt = 0; jt < 8; ++jt) {
                float p = __expf(accs[jt][e] - mnew[e]);
                sum += p;
                Ps[wave][lh * 8 + e][jt * 16 + ll] = (bf16)p;
            }
#pragma unroll
            for (int m = 1; m < 16; m <<= 1) sum += __shfl_xor(sum, m);
            const float corr = __expf(mrow[e] - mnew[e]);
            lrow[e] = lrow[e] * corr + sum;
            mrow[e] = mnew[e];
#pragma unroll
            for (int j = 0; j < 4; ++j) acco[j][e] *= corr;
        }

        // ---- O += P V  (per-wave LDS RAW is in-order on DScnt) ----
#pragma unroll
        for (int j = 0; j < 4; ++j)
#pragma unroll
            for (int kc = 0; kc < 4; ++kc) {
                bf16x16 a = *(const bf16x16*)&Ps[wave][ll][kc * 32 + lh * 16];
                bf16x16 b = *(const bf16x16*)&VTs[j * 16 + ll][kc * 32 + lh * 16];
                acco[j] = __builtin_amdgcn_wmma_f32_16x16x32_bf16(
                    false, a, false, b, (short)0, acco[j], false, false);
            }
    }

    // ---- write O = acc / l ----
#pragma unroll
    for (int j = 0; j < 4; ++j)
#pragma unroll
        for (int e = 0; e < 8; ++e) {
            const int row = q0 + wave * 16 + lh * 8 + e;
            const int d   = j * 16 + ll;
            O[base + (size_t)row * HDIM + d] = (bf16)(acco[j][e] / lrow[e]);
        }
}

// ---------------------------------------------------------------------------
// Host orchestration
// ---------------------------------------------------------------------------
extern "C" void kernel_launch(void* const* d_in, const int* in_sizes, int n_in,
                              void* d_out, int out_size, void* d_ws, size_t ws_size,
                              hipStream_t stream) {
    const float* x    = (const float*)d_in[0];
    const float* ln0g = (const float*)d_in[1];
    const float* ln0b = (const float*)d_in[2];
    const float* wq   = (const float*)d_in[3];
    const float* bq   = (const float*)d_in[4];
    const float* wk   = (const float*)d_in[5];
    const float* bk   = (const float*)d_in[6];
    const float* wv   = (const float*)d_in[7];
    const float* bv   = (const float*)d_in[8];
    const float* wo   = (const float*)d_in[9];
    const float* bo   = (const float*)d_in[10];
    const float* ln1g = (const float*)d_in[11];
    const float* ln1b = (const float*)d_in[12];
    const float* w1   = (const float*)d_in[13];
    const float* b1   = (const float*)d_in[14];
    const float* w2   = (const float*)d_in[15];
    const float* b2   = (const float*)d_in[16];
    float* out = (float*)d_out;

    char* p = (char*)d_ws;
    auto carve = [&](size_t bytes) -> char* {
        char* r = p;
        p += (bytes + 255) & ~(size_t)255;
        return r;
    };
    bf16*  wq_t = (bf16*)carve((size_t)HDIM * HDIM * 2);
    bf16*  wk_t = (bf16*)carve((size_t)HDIM * HDIM * 2);
    bf16*  wv_t = (bf16*)carve((size_t)HDIM * HDIM * 2);
    bf16*  wo_t = (bf16*)carve((size_t)HDIM * HDIM * 2);
    bf16*  w1_t = (bf16*)carve((size_t)MLPD * HDIM * 2);
    bf16*  w2_t = (bf16*)carve((size_t)HDIM * MLPD * 2);
    bf16*  y0   = (bf16*)carve((size_t)MTOK * HDIM * 2);   // LN0 out
    bf16*  qb   = (bf16*)carve((size_t)MTOK * HDIM * 2);
    bf16*  kb   = (bf16*)carve((size_t)MTOK * HDIM * 2);
    bf16*  vtb  = (bf16*)carve((size_t)MTOK * HDIM * 2);   // V, pre-transposed [b][h][d][key]
    bf16*  ab   = (bf16*)carve((size_t)MTOK * HDIM * 2);   // attention out
    float* x1   = (float*)carve((size_t)MTOK * HDIM * 4);  // residual 1
    bf16*  y1   = (bf16*)carve((size_t)MTOK * HDIM * 2);   // LN1 out
    bf16*  hb   = (bf16*)carve((size_t)MTOK * MLPD * 2);   // GELU(mlp1) out

    // --- weight convert/transpose (bf16, N x K) ---
    transpose_to_bf16<<<dim3(HDIM / 32, HDIM / 32), 256, 0, stream>>>(wq, wq_t, HDIM, HDIM);
    transpose_to_bf16<<<dim3(HDIM / 32, HDIM / 32), 256, 0, stream>>>(wk, wk_t, HDIM, HDIM);
    transpose_to_bf16<<<dim3(HDIM / 32, HDIM / 32), 256, 0, stream>>>(wv, wv_t, HDIM, HDIM);
    transpose_to_bf16<<<dim3(HDIM / 32, HDIM / 32), 256, 0, stream>>>(wo, wo_t, HDIM, HDIM);
    transpose_to_bf16<<<dim3(MLPD / 32, HDIM / 32), 256, 0, stream>>>(w1, w1_t, HDIM, MLPD);
    transpose_to_bf16<<<dim3(HDIM / 32, MLPD / 32), 256, 0, stream>>>(w2, w2_t, MLPD, HDIM);

    // --- pre-LN attention ---
    layernorm_to_bf16<<<MTOK, 256, 0, stream>>>(x, ln0g, ln0b, y0);

    dim3 gProj(HDIM / 128, MTOK / 128);
    gemm_bf16_wmma<EPI_BIAS, bf16><<<gProj, 256, 0, stream>>>(y0, wq_t, bq, nullptr, qb, MTOK, HDIM, HDIM);
    gemm_bf16_wmma<EPI_BIAS, bf16><<<gProj, 256, 0, stream>>>(y0, wk_t, bk, nullptr, kb, MTOK, HDIM, HDIM);
    gemm_bf16_wmma<EPI_BIAS_VT, bf16><<<gProj, 256, 0, stream>>>(y0, wv_t, bv, nullptr, vtb, MTOK, HDIM, HDIM);

    attention_wmma<<<dim3(LSEQ / 64, NH, NB), 128, 0, stream>>>(qb, kb, vtb, ab);

    gemm_bf16_wmma<EPI_BIAS_RES, float><<<gProj, 256, 0, stream>>>(ab, wo_t, bo, x, x1, MTOK, HDIM, HDIM);

    // --- pre-LN MLP ---
    layernorm_to_bf16<<<MTOK, 256, 0, stream>>>(x1, ln1g, ln1b, y1);

    gemm_bf16_wmma<EPI_BIAS_GELU, bf16><<<dim3(MLPD / 128, MTOK / 128), 256, 0, stream>>>(
        y1, w1_t, b1, nullptr, hb, MTOK, MLPD, HDIM);

    gemm_bf16_wmma<EPI_BIAS_RES, float><<<dim3(HDIM / 128, MTOK / 128), 256, 0, stream>>>(
        hb, w2_t, b2, x1, out, MTOK, HDIM, MLPD);
}
